// Graph_Mixer_84653805404414
// MI455X (gfx1250) — matrix operand order repaired and checked by reading, not compiled, not run on hardware
//
#include <hip/hip_runtime.h>
#include <hip/hip_bf16.h>
#include <math.h>

// Problem constants (match reference)
#define BB 32
#define SS 64
#define NN 32
#define BS 2048          // B*S
#define STATE 256
#define OBS 128
#define HID 64
#define ALPHA 0.2f
#define NEGV -9e15f

typedef __attribute__((ext_vector_type(16))) _Float16 v16h;
typedef __attribute__((ext_vector_type(8)))  _Float16 v8h;
typedef __attribute__((ext_vector_type(8)))  float    v8f;

// ---- WMMA fragment loaders (wave32, v_wmma_f32_16x16x32_f16 layouts) ----
// A (16x32, f16) from row-major LDS matrix, row stride ld (halves).
// lane L: M = Mbase + L%16 ; elems 0..7 -> K = kk + 8*(L/16) + e
//                            elems 8..15 -> K = kk + 16 + 8*(L/16) + (e-8)
__device__ __forceinline__ v16h frag_a(const _Float16* m, int ld, int Mbase, int kk) {
    int lane = threadIdx.x & 31;
    int g = lane >> 4;
    const _Float16* p = m + (Mbase + (lane & 15)) * ld + kk + 8 * g;
    v8h lo = *(const v8h*)p;
    v8h hi = *(const v8h*)(p + 16);
    v16h r;
#pragma unroll
    for (int e = 0; e < 8; ++e) { r[e] = lo[e]; r[e + 8] = hi[e]; }
    return r;
}

// B (32x16, f16) from N-major (transposed) LDS matrix mT[N][ld].
// lane L: N = Nbase + L%16 ; elems 0..7 -> K = kk + 8*(L/16) + e, elems 8..15 -> K = kk + 16 + 8*(L/16) + (e-8)
__device__ __forceinline__ v16h frag_b(const _Float16* mT, int ld, int Nbase, int kk) {
    int lane = threadIdx.x & 31;
    int g = lane >> 4;
    const _Float16* p = mT + (Nbase + (lane & 15)) * ld + kk + 8 * g;
    v8h lo = *(const v8h*)p;
    v8h hi = *(const v8h*)(p + 16);
    v16h r;
#pragma unroll
    for (int e = 0; e < 8; ++e) { r[e] = lo[e]; r[e + 8] = hi[e]; }
    return r;
}

// ============================================================================
// Kernel 1: per (b,s) — hypernet layer-1, h = obs @ W (WMMA), ei/ej
// ============================================================================
__global__ __launch_bounds__(256) void k1_layer1(
    const float* __restrict__ states, const float* __restrict__ obs,
    const float* __restrict__ w1, const float* __restrict__ b1,
    const float* __restrict__ w2, const float* __restrict__ b2,
    const float* __restrict__ aw1, const float* __restrict__ ab1,
    const float* __restrict__ aw2, const float* __restrict__ ab2,
    float* __restrict__ ws_h, float* __restrict__ ws_ei, float* __restrict__ ws_ej)
{
    __shared__ float s_state[STATE];
    __shared__ float s_t1[32];
    __shared__ float s_t1b[32];
    __shared__ __align__(16) _Float16 s_WT[HID * OBS];   // [o][f], 16 KB
    __shared__ float s_a[2 * HID];
    __shared__ __align__(16) _Float16 s_obs[NN * OBS];   // [n][f], 8 KB
    __shared__ float s_h[NN * HID];                      // 8 KB

    const int bs = blockIdx.x;
    const int t  = threadIdx.x;

    s_state[t] = states[bs * STATE + t];
    __syncthreads();

    if (t < 32) {                       // t1 = relu(states @ w1 + b1)
        float acc = b1[t];
#pragma unroll 8
        for (int s = 0; s < STATE; ++s) acc += s_state[s] * w1[s * 32 + t];
        s_t1[t] = fmaxf(acc, 0.f);
    } else if (t < 64) {                // t1b for attention MLP
        int k = t - 32;
        float acc = ab1[k];
#pragma unroll 8
        for (int s = 0; s < STATE; ++s) acc += s_state[s] * aw1[s * 32 + k];
        s_t1b[k] = fmaxf(acc, 0.f);
    }
    __syncthreads();

    // W = abs(t1 @ w2 + b2), reshape [128][64]; store transposed [o][f] as f16
#pragma unroll 4
    for (int it = 0; it < 32; ++it) {
        int idx = t + it * 256;          // idx = f*64 + o
        float acc = b2[idx];
#pragma unroll
        for (int k = 0; k < 32; ++k) acc += s_t1[k] * w2[k * 8192 + idx];
        s_WT[(idx & 63) * OBS + (idx >> 6)] = (_Float16)fabsf(acc);
    }
    if (t < 128) {                      // a = abs(t1b @ aw2 + ab2)
        float acc = ab2[t];
#pragma unroll
        for (int k = 0; k < 32; ++k) acc += s_t1b[k] * aw2[k * 128 + t];
        s_a[t] = fabsf(acc);
    }
#pragma unroll
    for (int it = 0; it < 16; ++it) {   // obs tile -> f16 LDS
        int idx = t + it * 256;
        s_obs[idx] = (_Float16)obs[bs * (NN * OBS) + idx];
    }
    __syncthreads();

    // h(32x64) = obs(32x128) @ W(128x64): 8 waves x 1 tile, 4 k-steps
    const int wave = t >> 5;
    const int Mb = (wave >> 2) * 16;
    const int Nb = (wave & 3) * 16;
    v8f c = {};
#pragma unroll
    for (int kk = 0; kk < OBS; kk += 32) {
        v16h av = frag_a(s_obs, OBS, Mb, kk);
        v16h bv = frag_b(s_WT, OBS, Nb, kk);
        c = __builtin_amdgcn_wmma_f32_16x16x32_f16(false, av, false, bv,
                                                   (short)0, c, false, false);
    }
    {
        const int lane = t & 31, g = lane >> 4, col = Nb + (lane & 15);
#pragma unroll
        for (int r = 0; r < 8; ++r) {
            int row = Mb + r + 8 * g;
            float v = c[r];
            s_h[row * HID + col] = v;
            ws_h[bs * (NN * HID) + row * HID + col] = v;
        }
    }
    __syncthreads();

    if (t < 32) {                       // ei[i] = h[i,:] . a[:64]
        float acc = 0.f;
#pragma unroll 8
        for (int o = 0; o < HID; ++o) acc += s_h[t * HID + o] * s_a[o];
        ws_ei[bs * NN + t] = acc;
    } else if (t < 64) {                // ej[j] = h[j,:] . a[64:]
        int j = t - 32;
        float acc = 0.f;
#pragma unroll 8
        for (int o = 0; o < HID; ++o) acc += s_h[j * HID + o] * s_a[HID + o];
        ws_ej[bs * NN + j] = acc;
    }
}

// ============================================================================
// Kernel 2: layer-1 attention softmax over the sequence axis (axis=1)
// one thread per (b,i,j); online max/sum pass then normalize pass
// ============================================================================
__global__ __launch_bounds__(256) void k2_softmax1(
    const float* __restrict__ ei, const float* __restrict__ ej,
    const int* __restrict__ adj, float* __restrict__ attn)
{
    int gid = blockIdx.x * 256 + threadIdx.x;   // < 32*32*32
    int b = gid >> 10, i = (gid >> 5) & 31, j = gid & 31;
    float m = -INFINITY, sum = 0.f;
    for (int s = 0; s < SS; ++s) {
        int bsn = (b * SS + s) * NN;
        float e = ei[bsn + i] + ej[bsn + j];
        e = e > 0.f ? e : ALPHA * e;
        float v = (adj[(bsn + i) * NN + j] > 0) ? e : NEGV;
        if (v > m) { sum = sum * __expf(m - v) + 1.f; m = v; }
        else       { sum += __expf(v - m); }
    }
    float inv = 1.f / sum;
    for (int s = 0; s < SS; ++s) {
        int bsn = (b * SS + s) * NN;
        float e = ei[bsn + i] + ej[bsn + j];
        e = e > 0.f ? e : ALPHA * e;
        float v = (adj[(bsn + i) * NN + j] > 0) ? e : NEGV;
        attn[(bsn + i) * NN + j] = __expf(v - m) * inv;
    }
}

// ============================================================================
// Kernel 3: per (b,s) — hp = attn @ h (WMMA), ELU -> h1 (LDS only);
//           layer-2 hypernet; h2 = h1 @ W2; a2
// ============================================================================
__global__ __launch_bounds__(256) void k3_layer2(
    const float* __restrict__ states,
    const float* __restrict__ ws_h, const float* __restrict__ ws_attn,
    const float* __restrict__ w1, const float* __restrict__ b1,
    const float* __restrict__ w2, const float* __restrict__ b2,
    const float* __restrict__ aw1, const float* __restrict__ ab1,
    const float* __restrict__ aw2, const float* __restrict__ ab2,
    float* __restrict__ ws_h2, float* __restrict__ ws_a2)
{
    __shared__ float s_state[STATE];
    __shared__ float s_t2[32];
    __shared__ float s_t2b[32];
    __shared__ __align__(16) _Float16 s_attn[NN * NN];   // [i][j]
    __shared__ __align__(16) _Float16 s_hT[HID * NN];    // [o][j]
    __shared__ float s_w2[HID];
    __shared__ float s_a2[2];
    __shared__ float s_h1[NN * HID];

    const int bs = blockIdx.x;
    const int t  = threadIdx.x;

    s_state[t] = states[bs * STATE + t];
#pragma unroll
    for (int it = 0; it < 4; ++it) {
        int idx = t + it * 256;
        s_attn[idx] = (_Float16)ws_attn[bs * (NN * NN) + idx];
    }
#pragma unroll
    for (int it = 0; it < 8; ++it) {
        int idx = t + it * 256;          // idx = j*64 + o
        s_hT[(idx & 63) * NN + (idx >> 6)] = (_Float16)ws_h[bs * (NN * HID) + idx];
    }
    __syncthreads();

    if (t < 32) {
        float acc = b1[t];
#pragma unroll 8
        for (int s = 0; s < STATE; ++s) acc += s_state[s] * w1[s * 32 + t];
        s_t2[t] = fmaxf(acc, 0.f);
    } else if (t < 64) {
        int k = t - 32;
        float acc = ab1[k];
#pragma unroll 8
        for (int s = 0; s < STATE; ++s) acc += s_state[s] * aw1[s * 32 + k];
        s_t2b[k] = fmaxf(acc, 0.f);
    }
    __syncthreads();

    if (t < HID) {                      // W2 = abs(t2 @ w2 + b2)  [64]
        float acc = b2[t];
#pragma unroll
        for (int k = 0; k < 32; ++k) acc += s_t2[k] * w2[k * HID + t];
        s_w2[t] = fabsf(acc);
    } else if (t < HID + 2) {           // a2 = abs(t2b @ aw2 + ab2) [2]
        int i2 = t - HID;
        float acc = ab2[i2];
#pragma unroll
        for (int k = 0; k < 32; ++k) acc += s_t2b[k] * aw2[k * 2 + i2];
        s_a2[i2] = fabsf(acc);
    }

    // hp(32x64) = attn(32x32) @ h(32x64): single 16x16x32 step per wave
    const int wave = t >> 5;
    const int Mb = (wave >> 2) * 16;
    const int Nb = (wave & 3) * 16;
    v16h av = frag_a(s_attn, NN, Mb, 0);
    v16h bv = frag_b(s_hT, NN, Nb, 0);
    v8f c = {};
    c = __builtin_amdgcn_wmma_f32_16x16x32_f16(false, av, false, bv,
                                               (short)0, c, false, false);
    {
        const int lane = t & 31, g = lane >> 4, col = Nb + (lane & 15);
#pragma unroll
        for (int r = 0; r < 8; ++r) {
            int row = Mb + r + 8 * g;
            float x = c[r];
            s_h1[row * HID + col] = x > 0.f ? x : (__expf(x) - 1.f);  // ELU
        }
    }
    __syncthreads();

    if (t < 32) {                       // h2[i] = h1[i,:] . W2
        float acc = 0.f;
#pragma unroll 8
        for (int o = 0; o < HID; ++o) acc += s_h1[t * HID + o] * s_w2[o];
        ws_h2[bs * NN + t] = acc;
    } else if (t == 32) {
        ws_a2[bs * 2 + 0] = s_a2[0];
        ws_a2[bs * 2 + 1] = s_a2[1];
    }
}

// ============================================================================
// Kernel 4: layer-2 attention softmax over the sequence axis
// ============================================================================
__global__ __launch_bounds__(256) void k4_softmax2(
    const float* __restrict__ h2, const float* __restrict__ a2,
    const int* __restrict__ adj, float* __restrict__ attn2)
{
    int gid = blockIdx.x * 256 + threadIdx.x;
    int b = gid >> 10, i = (gid >> 5) & 31, j = gid & 31;
    float m = -INFINITY, sum = 0.f;
    for (int s = 0; s < SS; ++s) {
        int base = b * SS + s;
        float e = h2[base * NN + i] * a2[base * 2] + h2[base * NN + j] * a2[base * 2 + 1];
        e = e > 0.f ? e : ALPHA * e;
        float v = (adj[(base * NN + i) * NN + j] > 0) ? e : NEGV;
        if (v > m) { sum = sum * __expf(m - v) + 1.f; m = v; }
        else       { sum += __expf(v - m); }
    }
    float inv = 1.f / sum;
    for (int s = 0; s < SS; ++s) {
        int base = b * SS + s;
        float e = h2[base * NN + i] * a2[base * 2] + h2[base * NN + j] * a2[base * 2 + 1];
        e = e > 0.f ? e : ALPHA * e;
        float v = (adj[(base * NN + i) * NN + j] > 0) ? e : NEGV;
        attn2[(base * NN + i) * NN + j] = __expf(v - m) * inv;
    }
}

// ============================================================================
// Kernel 5: gat_op[bs,i] = sum_j attn2[bs,i,j] * h2[bs,j]
// ============================================================================
__global__ __launch_bounds__(256) void k5_gat(
    const float* __restrict__ attn2, const float* __restrict__ h2,
    float* __restrict__ gat)
{
    int gid = blockIdx.x * 256 + threadIdx.x;   // < BS*NN
    int bs = gid >> 5, i = gid & 31;
    float acc = 0.f;
#pragma unroll 8
    for (int j = 0; j < NN; ++j)
        acc += attn2[bs * (NN * NN) + i * NN + j] * h2[bs * NN + j];
    gat[gid] = acc;
}

// ============================================================================
// Kernel 6a: per-column (agent) max & exp-sum over the 2048 rows
// ============================================================================
__global__ __launch_bounds__(256) void k6a_colstats(
    const float* __restrict__ gat, float* __restrict__ colmax, float* __restrict__ colsum)
{
    __shared__ float red[256];
    const int n = blockIdx.x, t = threadIdx.x;
    float m = -INFINITY;
    for (int r = t; r < BS; r += 256) m = fmaxf(m, gat[r * NN + n]);
    red[t] = m; __syncthreads();
    for (int o = 128; o > 0; o >>= 1) {
        if (t < o) red[t] = fmaxf(red[t], red[t + o]);
        __syncthreads();
    }
    float mx = red[0]; __syncthreads();
    float s = 0.f;
    for (int r = t; r < BS; r += 256) s += __expf(gat[r * NN + n] - mx);
    red[t] = s; __syncthreads();
    for (int o = 128; o > 0; o >>= 1) {
        if (t < o) red[t] += red[t + o];
        __syncthreads();
    }
    if (t == 0) { colmax[n] = mx; colsum[n] = red[0]; }
}

// ============================================================================
// Kernel 6b: q_tot[r] = sum_n qs[r,n] * softmax0(gat)[r,n]
// ============================================================================
__global__ __launch_bounds__(256) void k6b_qtot(
    const float* __restrict__ qs, const float* __restrict__ gat,
    const float* __restrict__ colmax, const float* __restrict__ colsum,
    float* __restrict__ out)
{
    __shared__ float cm[NN];
    __shared__ float cs[NN];
    int t = threadIdx.x;
    if (t < NN) cm[t] = colmax[t];
    else if (t < 2 * NN) cs[t - NN] = colsum[t - NN];
    __syncthreads();
    int r = blockIdx.x * 256 + t;
    float acc = 0.f;
#pragma unroll
    for (int n = 0; n < NN; ++n)
        acc += qs[r * NN + n] * __expf(gat[r * NN + n] - cm[n]) / cs[n];
    out[r] = acc;
}

// ============================================================================
extern "C" void kernel_launch(void* const* d_in, const int* in_sizes, int n_in,
                              void* d_out, int out_size, void* d_ws, size_t ws_size,
                              hipStream_t stream) {
    const float* agent_qs = (const float*)d_in[0];
    const float* states   = (const float*)d_in[1];
    const float* obs      = (const float*)d_in[2];
    const int*   adj      = (const int*)d_in[3];
    const float* p1w1  = (const float*)d_in[4];
    const float* p1b1  = (const float*)d_in[5];
    const float* p1w2  = (const float*)d_in[6];
    const float* p1b2  = (const float*)d_in[7];
    const float* p1aw1 = (const float*)d_in[8];
    const float* p1ab1 = (const float*)d_in[9];
    const float* p1aw2 = (const float*)d_in[10];
    const float* p1ab2 = (const float*)d_in[11];
    const float* p2w1  = (const float*)d_in[12];
    const float* p2b1  = (const float*)d_in[13];
    const float* p2w2  = (const float*)d_in[14];
    const float* p2b2  = (const float*)d_in[15];
    const float* p2aw1 = (const float*)d_in[16];
    const float* p2ab1 = (const float*)d_in[17];
    const float* p2aw2 = (const float*)d_in[18];
    const float* p2ab2 = (const float*)d_in[19];
    float* out = (float*)d_out;

    // Workspace layout (floats)
    float* w = (float*)d_ws;
    float* ws_h     = w;                          // BS*32*64   = 4,194,304
    float* ws_ei    = ws_h     + (size_t)BS * NN * HID;   // BS*32
    float* ws_ej    = ws_ei    + (size_t)BS * NN;
    float* ws_attn  = ws_ej    + (size_t)BS * NN;         // BS*32*32
    float* ws_attn2 = ws_attn  + (size_t)BS * NN * NN;
    float* ws_h2    = ws_attn2 + (size_t)BS * NN * NN;    // BS*32
    float* ws_a2    = ws_h2    + (size_t)BS * NN;         // BS*2
    float* ws_gat   = ws_a2    + (size_t)BS * 2;          // BS*32
    float* ws_cmax  = ws_gat   + (size_t)BS * NN;         // 32
    float* ws_csum  = ws_cmax  + NN;                      // 32

    k1_layer1<<<BS, 256, 0, stream>>>(states, obs,
                                      p1w1, p1b1, p1w2, p1b2,
                                      p1aw1, p1ab1, p1aw2, p1ab2,
                                      ws_h, ws_ei, ws_ej);
    k2_softmax1<<<(BB * NN * NN) / 256, 256, 0, stream>>>(ws_ei, ws_ej, adj, ws_attn);
    k3_layer2<<<BS, 256, 0, stream>>>(states, ws_h, ws_attn,
                                      p2w1, p2b1, p2w2, p2b2,
                                      p2aw1, p2ab1, p2aw2, p2ab2,
                                      ws_h2, ws_a2);
    k4_softmax2<<<(BB * NN * NN) / 256, 256, 0, stream>>>(ws_h2, ws_a2, adj, ws_attn2);
    k5_gat<<<(BS * NN) / 256, 256, 0, stream>>>(ws_attn2, ws_h2, ws_gat);
    k6a_colstats<<<NN, 256, 0, stream>>>(ws_gat, ws_cmax, ws_csum);
    k6b_qtot<<<BS / 256, 256, 0, stream>>>(agent_qs, ws_gat, ws_cmax, ws_csum, out);
}
